// IntegralTransform_2911987826756
// MI455X (gfx1250) — matrix operand order repaired and checked by reading, not compiled, not run on hardware
//
#include <hip/hip_runtime.h>
#include <hip/hip_bf16.h>
#include <math.h>

typedef __bf16 v16bf __attribute__((ext_vector_type(16)));
typedef float  v8f   __attribute__((ext_vector_type(8)));

#define KNBR 32
#define DF   64
#define HDIM 256
#define KPAD 96            // 70 -> 96 (3 x 32 k-steps)

#define W1T_HALVES (HDIM*KPAD)     // 24576
#define W2T_HALVES (HDIM*HDIM)     // 65536
#define W3T_HALVES (DF*HDIM)       // 16384
#define WS_HALVES  (W1T_HALVES + W2T_HALVES + W3T_HALVES)  // 106496

#define WAVES 4
#define QPW   8
#define BLOCKS 512          // 512*4 waves * 8 queries = 16384 = M

// per-wave LDS scratch: agg(16x96) + h1(16x256) + h2(16x256) bf16 + attn(32 f32)
#define WAVE_BYTES (16*KPAD*2 + 16*HDIM*2 + 16*HDIM*2 + 32*4)   // 19584
#define SMEM_BYTES (WS_HALVES*2 + 576*4 + WAVES*WAVE_BYTES)     // 293632

static_assert(SMEM_BYTES <= 320*1024, "exceeds WGP LDS");

__device__ __forceinline__ unsigned short f2bf(float x) {
  unsigned int u = __float_as_uint(x);
  unsigned int r = (u + 0x7FFFu + ((u >> 16) & 1u)) >> 16;   // RNE
  return (unsigned short)r;
}
__device__ __forceinline__ float bf2f(unsigned short h) {
  return __uint_as_float(((unsigned int)h) << 16);
}
__device__ __forceinline__ float gelu_exact(float v) {
  return 0.5f * v * (1.0f + erff(v * 0.70710678118654752f));
}

// A fragment (16x32 bf16, row-major in LDS, rowStride in halves).
// lane<16 : row=lane,    K = kbase+{0..7}, kbase+{16..23}
// lane>=16: row=lane-16, K = kbase+{8..15}, kbase+{24..31}
__device__ __forceinline__ v16bf load_frag_A(const unsigned short* base,
                                             int rowStride, int lane, int kbase) {
  const int half = lane >> 4, r = lane & 15;
  const unsigned short* p = base + r * rowStride + kbase + half * 8;
  union { v16bf v; uint4 q[2]; } f;
  f.q[0] = *reinterpret_cast<const uint4*>(p);
  f.q[1] = *reinterpret_cast<const uint4*>(p + 16);
  return f.v;
}

// B fragment (32x16 bf16) from transposed weights wT[N][K] (rowStride in halves).
// lane<16 : col=lane,    K = kbase+{0..7}, kbase+{16..23}
// lane>=16: col=lane-16, K = kbase+{8..15}, kbase+{24..31}
__device__ __forceinline__ v16bf load_frag_B(const unsigned short* tile,
                                             int rowStride, int lane, int kbase) {
  const int half = lane >> 4, n = lane & 15;
  const unsigned short* p = tile + n * rowStride + kbase + half * 8;
  union { v16bf v; uint4 q[2]; } f;
  f.q[0] = *reinterpret_cast<const uint4*>(p);
  f.q[1] = *reinterpret_cast<const uint4*>(p + 16);
  return f.v;
}

// ---- pre-pass: bf16-convert + transpose weights into workspace ----
__global__ void it_prep_kernel(const float* __restrict__ W1,
                               const float* __restrict__ W2,
                               const float* __restrict__ W3,
                               unsigned short* __restrict__ wsT) {
  int i = blockIdx.x * 256 + threadIdx.x;
  if (i < W1T_HALVES) {                       // W1T[n][k], k padded to 96
    int n = i / KPAD, k = i % KPAD;
    wsT[i] = (k < 70) ? f2bf(W1[k * HDIM + n]) : (unsigned short)0;
  } else if (i < W1T_HALVES + W2T_HALVES) {   // W2T[n][k]
    int j = i - W1T_HALVES;
    int n = j / HDIM, k = j % HDIM;
    wsT[i] = f2bf(W2[k * HDIM + n]);
  } else if (i < WS_HALVES) {                 // W3T[n][k]
    int j = i - (W1T_HALVES + W2T_HALVES);
    int n = j / HDIM, k = j % HDIM;
    wsT[i] = f2bf(W3[k * DF + n]);
  }
}

// ---- main kernel: one query (32 edges) per wave iteration ----
__global__ __launch_bounds__(WAVES * 32) void it_main_kernel(
    const float* __restrict__ y, const float* __restrict__ x,
    const float* __restrict__ f_y, const int* __restrict__ nbr,
    const float* __restrict__ b1, const float* __restrict__ b2,
    const float* __restrict__ b3, const unsigned short* __restrict__ wsT,
    float* __restrict__ out) {
  extern __shared__ __align__(16) unsigned char smem[];
  unsigned short* w1T = (unsigned short*)smem;
  unsigned short* w2T = w1T + W1T_HALVES;
  unsigned short* w3T = w2T + W2T_HALVES;
  float* sbias = (float*)(smem + (size_t)WS_HALVES * 2);  // b1|b2|b3 = 576 f32
  unsigned char* waveMem = smem + (size_t)WS_HALVES * 2 + 576 * 4;

  const int tid = threadIdx.x;

  // cooperative load: transposed bf16 weights + biases -> LDS
  {
    const uint4* src = (const uint4*)wsT;
    uint4* dst = (uint4*)smem;
    const int n16 = WS_HALVES / 8;            // 13312 x 16B
    for (int i = tid; i < n16; i += WAVES * 32) dst[i] = src[i];
    for (int i = tid; i < 576; i += WAVES * 32) {
      float v;
      if (i < 256)      v = b1[i];
      else if (i < 512) v = b2[i - 256];
      else              v = b3[i - 512];
      sbias[i] = v;
    }
  }
  __syncthreads();

  const int wave = tid >> 5;
  const int lane = tid & 31;
  const int half = lane >> 4;
  const int l16  = lane & 15;
  const int rbase = 8 * half;

  unsigned char* wb = waveMem + wave * WAVE_BYTES;
  unsigned short* agg = (unsigned short*)wb;            // 16 x 96 bf16
  unsigned short* h1  = agg + 16 * KPAD;                // 16 x 256 bf16
  unsigned short* h2  = h1 + 16 * HDIM;                 // 16 x 256 bf16
  float* attnW = (float*)(h2 + 16 * HDIM);              // 32 f32

  // zero agg once (pad columns 70..95 stay zero forever)
  for (int i = lane; i < 16 * KPAD; i += 32) agg[i] = 0;

  const int slot = blockIdx.x * WAVES + wave;           // 0..2047

  for (int q = 0; q < QPW; ++q) {
    const int m = slot * QPW + q;                       // query row
    __syncthreads();

    // ---- stage A: gather edge data, cosine score, 32-wide softmax ----
    const int e   = m * KNBR + lane;
    const int idx = nbr[e];
    const float ry0 = y[3 * idx + 0], ry1 = y[3 * idx + 1], ry2 = y[3 * idx + 2];
    const float sx0 = x[3 * m + 0],  sx1 = x[3 * m + 1],  sx2 = x[3 * m + 2];
    const float nr = sqrtf(ry0 * ry0 + ry1 * ry1 + ry2 * ry2);
    const float ns = sqrtf(sx0 * sx0 + sx1 * sx1 + sx2 * sx2);
    const float inv = (1.0f / fmaxf(nr, 1e-12f)) * (1.0f / fmaxf(ns, 1e-12f));
    const float score = (ry0 * sx0 + ry1 * sx1 + ry2 * sx2) * inv;
    float mx = score;
    for (int off = 16; off > 0; off >>= 1) mx = fmaxf(mx, __shfl_xor(mx, off, 32));
    const float ex = expf(score - mx);
    float sm = ex;
    for (int off = 16; off > 0; off >>= 1) sm += __shfl_xor(sm, off, 32);
    attnW[lane] = ex / sm;

    float outv[4] = {0.0f, 0.0f, 0.0f, 0.0f};

    for (int hh = 0; hh < 2; ++hh) {                    // two 16-edge row tiles
      __syncthreads();
      // ---- fill agg tile (rows = edges hh*16..hh*16+15) ----
      if (half == hh) {
        unsigned short* arow = agg + l16 * KPAD;
        arow[0] = f2bf(ry0); arow[1] = f2bf(ry1); arow[2] = f2bf(ry2);
        arow[3] = f2bf(sx0); arow[4] = f2bf(sx1); arow[5] = f2bf(sx2);
        const float4* fp = (const float4*)(f_y + (size_t)idx * DF);
#pragma unroll
        for (int j4 = 0; j4 < 16; ++j4) {
          float4 v = fp[j4];
          arow[6 + 4 * j4 + 0] = f2bf(v.x);
          arow[6 + 4 * j4 + 1] = f2bf(v.y);
          arow[6 + 4 * j4 + 2] = f2bf(v.z);
          arow[6 + 4 * j4 + 3] = f2bf(v.w);
        }
      }
      __syncthreads();

      // ---- GEMM1: [16x96] @ W1T -> gelu -> h1 [16x256] ----
      for (int t = 0; t < 16; ++t) {
        const int col = l16 + 16 * t;
        v8f acc;
        const float bv = sbias[col];
#pragma unroll
        for (int j = 0; j < 8; ++j) acc[j] = bv;
#pragma unroll
        for (int kk = 0; kk < 3; ++kk) {
          v16bf a = load_frag_A(agg, KPAD, lane, kk * 32);
          v16bf b = load_frag_B(w1T + (t * 16) * KPAD, KPAD, lane, kk * 32);
          acc = __builtin_amdgcn_wmma_f32_16x16x32_bf16(false, a, false, b,
                                                        (short)0, acc, false, false);
        }
#pragma unroll
        for (int j = 0; j < 8; ++j)
          h1[(rbase + j) * HDIM + col] = f2bf(gelu_exact(acc[j]));
      }
      __syncthreads();

      // ---- GEMM2: [16x256] @ W2T -> gelu -> h2 [16x256] ----
      for (int t = 0; t < 16; ++t) {
        const int col = l16 + 16 * t;
        v8f acc;
        const float bv = sbias[256 + col];
#pragma unroll
        for (int j = 0; j < 8; ++j) acc[j] = bv;
#pragma unroll
        for (int kk = 0; kk < 8; ++kk) {
          v16bf a = load_frag_A(h1, HDIM, lane, kk * 32);
          v16bf b = load_frag_B(w2T + (t * 16) * HDIM, HDIM, lane, kk * 32);
          acc = __builtin_amdgcn_wmma_f32_16x16x32_bf16(false, a, false, b,
                                                        (short)0, acc, false, false);
        }
#pragma unroll
        for (int j = 0; j < 8; ++j)
          h2[(rbase + j) * HDIM + col] = f2bf(gelu_exact(acc[j]));
      }
      __syncthreads();

      // ---- GEMM3: [16x256] @ W3T -> * f_y * attn -> row-reduce ----
      for (int t = 0; t < 4; ++t) {
        const int col = l16 + 16 * t;
        v8f acc;
        const float bv = sbias[512 + col];
#pragma unroll
        for (int j = 0; j < 8; ++j) acc[j] = bv;
#pragma unroll
        for (int kk = 0; kk < 8; ++kk) {
          v16bf a = load_frag_A(h2, HDIM, lane, kk * 32);
          v16bf b = load_frag_B(w3T + (t * 16) * HDIM, HDIM, lane, kk * 32);
          acc = __builtin_amdgcn_wmma_f32_16x16x32_bf16(false, a, false, b,
                                                        (short)0, acc, false, false);
        }
        float partial = 0.0f;
#pragma unroll
        for (int j = 0; j < 8; ++j) {
          const int r = rbase + j;
          const float infe = bf2f(agg[r * KPAD + 6 + col]);   // f_y reused from agg
          partial += acc[j] * attnW[hh * 16 + r] * infe;
        }
        partial += __shfl_xor(partial, 16, 32);               // add other row-half
        outv[t] += partial;
      }
    }

    // ---- store 64 outputs for query m (each col once) ----
    {
      float* op = out + (size_t)m * DF;
      const int t0 = 2 * half;                                // lanes 0-15: t=0,1; 16-31: t=2,3
      op[l16 + 16 * (t0 + 0)] = outv[t0 + 0];
      op[l16 + 16 * (t0 + 1)] = outv[t0 + 1];
    }
  }
}

extern "C" void kernel_launch(void* const* d_in, const int* in_sizes, int n_in,
                              void* d_out, int out_size, void* d_ws, size_t ws_size,
                              hipStream_t stream) {
  const float* y  = (const float*)d_in[0];
  const float* x  = (const float*)d_in[1];
  const float* fy = (const float*)d_in[2];
  const int*   nb = (const int*)d_in[3];
  // d_in[4] = neighbors_row_splits: uniform K=32 per the reference setup
  const float* W1 = (const float*)d_in[5];
  const float* b1 = (const float*)d_in[6];
  const float* W2 = (const float*)d_in[7];
  const float* b2 = (const float*)d_in[8];
  const float* W3 = (const float*)d_in[9];
  const float* b3 = (const float*)d_in[10];
  unsigned short* wsT = (unsigned short*)d_ws;   // 212992 B of scratch used
  float* out = (float*)d_out;

  it_prep_kernel<<<(WS_HALVES + 255) / 256, 256, 0, stream>>>(W1, W2, W3, wsT);

  hipFuncSetAttribute(reinterpret_cast<const void*>(it_main_kernel),
                      hipFuncAttributeMaxDynamicSharedMemorySize, SMEM_BYTES);
  it_main_kernel<<<BLOCKS, WAVES * 32, SMEM_BYTES, stream>>>(y, x, fy, nb,
                                                             b1, b2, b3, wsT, out);
}
